// SDFDecoder_85993835200981
// MI455X (gfx1250) — matrix operand order repaired and checked by reading, not compiled, not run on hardware
//
#include <hip/hip_runtime.h>
#include <hip/hip_bf16.h>
#include <math.h>

// ---------------------------------------------------------------------------
// SDF decoder (KNN + residual-concat MLP) for MI455X / gfx1250.
// All matmuls on v_wmma_f32_16x16x32_bf16 (wave32 WMMA), weights pre-swizzled
// to B-fragment layout in d_ws (L2-resident), activations in-place in LDS.
// ---------------------------------------------------------------------------

typedef __bf16 bf16;
typedef __attribute__((ext_vector_type(16))) bf16  v16bf;
typedef __attribute__((ext_vector_type(8)))  float v8f;

#define NPTS    65536
#define INROW   1539          // 3 + 3*512 floats per point
#define HDIM    512
#define F0DIM   1563          // 27 + 1536
#define MROWS   32            // points per workgroup (two WMMA M tiles)
#define X_ACT   0             // activation x (512) base inside a row
#define X_AUG   512           // x_ (1568) base inside a row
#define ROWLEN  2184          // 512 + 1568 + pad (4368B/row: 16B-aligned)
#define NT      32            // 512 / 16 N-tiles
#define KT_FIRST 49           // 1568 / 32
#define KT_IN    65           // 2080 / 32
#define KT_HID   16           // 512  / 32

#define WF_ELEMS  ((size_t)KT_FIRST * NT * 512)   //  802816
#define WIN_ELEMS ((size_t)KT_IN    * NT * 512)   // 1064960
#define WH_ELEMS  ((size_t)KT_HID   * NT * 512)   //  262144
// ws requirement: (WF + 3*WIN + 8*WH) * 2B ~= 11.7 MB

// ---------------------------------------------------------------------------
// Weight prep: fp32 [K][512] row-major -> bf16 WMMA-B swizzled tiles.
// Tile (kt,nt): lane L stores 16 contiguous bf16 at tile*512 + L*16:
//   K = kt*32 + e + (L>=16 ? 16 : 0),   N = nt*16 + (L&15).
// K >= Ksrc rows are zero (K padding).
// ---------------------------------------------------------------------------
__global__ void wswizzle_bf16_kernel(const float* __restrict__ src,
                                     bf16* __restrict__ dst,
                                     int Ksrc, int ktiles) {
  int t = blockIdx.x * blockDim.x + threadIdx.x;
  int total = ktiles * NT * 32;
  if (t >= total) return;
  int lane = t & 31;
  int tile = t >> 5;                       // kt*NT + nt
  int kt   = tile >> 5;
  int nt   = tile & 31;
  int kb   = kt * 32 + ((lane >= 16) ? 16 : 0);
  int n    = nt * 16 + (lane & 15);
  bf16* o  = dst + (size_t)tile * 512 + lane * 16;
#pragma unroll
  for (int e = 0; e < 16; ++e) {
    int k   = kb + e;
    float v = (k < Ksrc) ? src[(size_t)k * HDIM + n] : 0.0f;
    o[e] = (bf16)v;
  }
}

// ---------------------------------------------------------------------------
// One dense layer, in-place: x = relu(A @ W + b).
//   K-tile kt:  kt < splitKt -> activation region (K = kt*32)
//               else         -> x_ region (K = (kt-splitKt)*32)
// All waves read in the K loop, barrier, then overwrite the activation
// region, barrier.  Wave owns M-tile (wave&1) and N-tiles (wave>>1)*8..+7.
// ---------------------------------------------------------------------------
__device__ __forceinline__ void mlp_layer(bf16* smem, int splitKt, int ktiles,
                                          const bf16* __restrict__ W,
                                          const float* __restrict__ bias,
                                          int lane, int wave) {
  const int mtile = wave & 1;
  const int nt0   = (wave >> 1) * 8;
  const int m     = lane & 15;
  const int hi    = lane >> 4;             // 0: lanes 0-15, 1: lanes 16-31

  v8f acc[8] = {};

  for (int kt = 0; kt < ktiles; ++kt) {
    int kbase = (kt < splitKt) ? (X_ACT + kt * 32)
                               : (X_AUG + (kt - splitKt) * 32);
    // A fragment (16-bit A 16x32 layout): two 16B chunks per lane,
    // K offsets hi*16 and hi*16 + 8 (contiguous, matches B swizzle).
    const bf16* ap = smem + (mtile * 16 + m) * ROWLEN + kbase + hi * 16;
    v16bf a;
    ((uint4*)&a)[0] = *(const uint4*)(ap);
    ((uint4*)&a)[1] = *(const uint4*)(ap + 8);

    int ktn = (kt + 1 < ktiles) ? (kt + 1) : kt;     // prefetch target
#pragma unroll
    for (int n = 0; n < 8; ++n) {
      const bf16* wp = W + ((size_t)(kt * NT + nt0 + n)) * 512 + lane * 16;
      // prefetch next K-tile's fragment for this N-tile (global_prefetch_b8)
      __builtin_prefetch(W + ((size_t)(ktn * NT + nt0 + n)) * 512 + lane * 16,
                         0, 3);
      v16bf b;
      ((uint4*)&b)[0] = ((const uint4*)wp)[0];
      ((uint4*)&b)[1] = ((const uint4*)wp)[1];
      acc[n] = __builtin_amdgcn_wmma_f32_16x16x32_bf16(
          /*neg_a=*/false, a, /*neg_b=*/false, b,
          /*c_mod=*/(short)0, acc[n], /*reuse_a=*/false, /*reuse_b=*/false);
    }
  }

  __syncthreads();   // all reads of the activation region complete

  // bias + relu + bf16 writeback (C/D layout: VGPR r -> M = r + hi*8)
#pragma unroll
  for (int n = 0; n < 8; ++n) {
    int ncol = (nt0 + n) * 16 + m;
    float bv = bias[ncol];
#pragma unroll
    for (int r = 0; r < 8; ++r) {
      int rowM = mtile * 16 + r + hi * 8;
      float v = acc[n][r] + bv;
      v = v > 0.0f ? v : 0.0f;
      smem[rowM * ROWLEN + X_ACT + ncol] = (bf16)v;
    }
  }

  __syncthreads();   // writes visible before the next layer reads
}

// ---------------------------------------------------------------------------
// Main kernel: 32 points per workgroup, 8 wave32 waves.
// ---------------------------------------------------------------------------
__global__ __launch_bounds__(256) void sdf_decoder_kernel(
    const float* __restrict__ cxyz,
    const bf16* __restrict__ wfirst, const bf16* __restrict__ win,
    const bf16* __restrict__ whid,
    const float* __restrict__ bfirst, const float* __restrict__ bin,
    const float* __restrict__ bhid,
    const float* __restrict__ wout, const float* __restrict__ bout,
    float* __restrict__ out) {
  __shared__ bf16 smem[MROWS * ROWLEN];    // 139,776 bytes

  const int tid  = threadIdx.x;
  const int lane = tid & 31;
  const int wave = tid >> 5;
  const int blk  = blockIdx.x;

  // ---------------- Phase A: KNN + build x_ (27 header + 1536 latent) ------
  for (int rr = 0; rr < 4; ++rr) {
    int row = wave * 4 + rr;
    const float* p = cxyz + (size_t)(blk * MROWS + row) * INROW;
    float x0 = p[0], y0 = p[1], z0 = p[2];
    bf16* xr = smem + row * ROWLEN + X_AUG;
    if (lane == 0) { xr[0] = (bf16)x0; xr[1] = (bf16)y0; xr[2] = (bf16)z0; }
    if (lane < 5)  { xr[F0DIM + lane] = (bf16)0.0f; }   // K-pad 1563..1567

    float d2l[16];
#pragma unroll
    for (int c = 0; c < 16; ++c) {
      int j = lane + 32 * c;
      float ax = p[3 + 3 * j], ay = p[4 + 3 * j], az = p[5 + 3 * j];
      float dx = x0 - ax, dy = y0 - ay, dz = z0 - az;
      d2l[c] = dx * dx + dy * dy + dz * dz;
      xr[27 + 3 * j]     = (bf16)ax;
      xr[27 + 3 * j + 1] = (bf16)ay;
      xr[27 + 3 * j + 2] = (bf16)az;
    }
    // top-8 by repeated wave-wide argmin (tie-break: smaller index)
#pragma unroll 1
    for (int i = 0; i < 8; ++i) {
      float bv = d2l[0]; int bc = 0;
#pragma unroll
      for (int c = 1; c < 16; ++c)
        if (d2l[c] < bv) { bv = d2l[c]; bc = c; }
      int bj = lane + 32 * bc;
      for (int off = 16; off; off >>= 1) {
        float ov = __shfl_xor(bv, off, 32);
        int   oj = __shfl_xor(bj, off, 32);
        if (ov < bv || (ov == bv && oj < bj)) { bv = ov; bj = oj; }
      }
      if ((bj & 31) == lane) d2l[bj >> 5] = 3.4e38f;   // knock out winner
      if (lane == 0) {                                  // gather sel -> header
        xr[3 + 3 * i]     = xr[27 + 3 * bj];
        xr[4 + 3 * i]     = xr[28 + 3 * bj];
        xr[5 + 3 * i]     = xr[29 + 3 * bj];
      }
    }
  }
  __syncthreads();

  // ---------------- MLP (all layers write activations in place) ------------
  // First layer: A entirely from x_ (splitKt = 0).
  mlp_layer(smem, /*splitKt=*/0, KT_FIRST, wfirst, bfirst, lane, wave);
  // Block 0 hidden layers
  for (int l = 0; l < 2; ++l)
    mlp_layer(smem, KT_HID, KT_HID, whid + (size_t)l * WH_ELEMS,
              bhid + l * HDIM, lane, wave);
  // Blocks 1..3: concat(x, x_) layer + 2 hidden layers
  for (int b = 1; b < 4; ++b) {
    mlp_layer(smem, /*splitKt=*/16, KT_IN,
              win + (size_t)(b - 1) * WIN_ELEMS, bin + (b - 1) * HDIM,
              lane, wave);
    for (int l = 0; l < 2; ++l) {
      int li = b * 2 + l;
      mlp_layer(smem, KT_HID, KT_HID, whid + (size_t)li * WH_ELEMS,
                bhid + li * HDIM, lane, wave);
    }
  }

  // ---------------- Output head: tanh(x @ W_out + b) -----------------------
  for (int rr = 0; rr < 4; ++rr) {
    int row = wave * 4 + rr;
    const bf16* xr = smem + row * ROWLEN + X_ACT;
    float s = 0.0f;
#pragma unroll
    for (int k = 0; k < 16; ++k)
      s += (float)xr[lane * 16 + k] * wout[lane * 16 + k];
    for (int off = 16; off; off >>= 1) s += __shfl_xor(s, off, 32);
    if (lane == 0) out[blk * MROWS + row] = tanhf(s + bout[0]);
  }
}

// ---------------------------------------------------------------------------
extern "C" void kernel_launch(void* const* d_in, const int* in_sizes, int n_in,
                              void* d_out, int out_size, void* d_ws,
                              size_t ws_size, hipStream_t stream) {
  const float* cxyz     = (const float*)d_in[0];
  const float* W_first  = (const float*)d_in[1];
  const float* b_first  = (const float*)d_in[2];
  const float* W_in     = (const float*)d_in[3];
  const float* b_in     = (const float*)d_in[4];
  const float* W_hidden = (const float*)d_in[5];
  const float* b_hidden = (const float*)d_in[6];
  const float* W_out    = (const float*)d_in[7];
  const float* b_out    = (const float*)d_in[8];
  float* out = (float*)d_out;

  bf16* wf = (bf16*)d_ws;                 // pre-swizzled bf16 weights in ws
  bf16* wi = wf + WF_ELEMS;
  bf16* wh = wi + 3 * WIN_ELEMS;

  // Weight conversion + swizzle (deterministic, re-runs every launch).
  {
    int total = KT_FIRST * NT * 32;
    wswizzle_bf16_kernel<<<(total + 255) / 256, 256, 0, stream>>>(
        W_first, wf, F0DIM, KT_FIRST);
  }
  for (int b = 0; b < 3; ++b) {
    int total = KT_IN * NT * 32;
    wswizzle_bf16_kernel<<<(total + 255) / 256, 256, 0, stream>>>(
        W_in + (size_t)b * 2075 * HDIM, wi + (size_t)b * WIN_ELEMS,
        2075, KT_IN);
  }
  for (int i = 0; i < 8; ++i) {
    int total = KT_HID * NT * 32;
    wswizzle_bf16_kernel<<<(total + 255) / 256, 256, 0, stream>>>(
        W_hidden + (size_t)i * HDIM * HDIM, wh + (size_t)i * WH_ELEMS,
        HDIM, KT_HID);
  }

  sdf_decoder_kernel<<<NPTS / MROWS, 256, 0, stream>>>(
      cxyz, wf, wi, wh, b_first, b_in, b_hidden, W_out, b_out, out);
}
